// HGNN_46351287058754
// MI455X (gfx1250) — matrix operand order repaired and checked by reading, not compiled, not run on hardware
//
#include <hip/hip_runtime.h>
#include <hip/hip_bf16.h>

typedef __attribute__((ext_vector_type(16))) _Float16 v16h;
typedef __attribute__((ext_vector_type(8)))  _Float16 v8h;
typedef __attribute__((ext_vector_type(8)))  float    v8f;

#define LRELU_ALPHA 0.2f
#define PANEL_K  256                    // K-panel staged in LDS per barrier
#define PSTRIDE  264                    // halves per B row in LDS (256 + 8 pad)
#define GEMM_LDS_BYTES (64 * PSTRIDE * 2)

__device__ __forceinline__ float act_apply(float v, int act) {
    if (act == 1) return v >= 0.f ? v : LRELU_ALPHA * v;
    if (act == 2) return tanhf(v);
    return v;
}

// ---------------------------------------------------------------------------
// f32 -> f16 bulk convert (weights), 4 elements/thread
// ---------------------------------------------------------------------------
__global__ __launch_bounds__(256) void f32_to_f16_kernel(
    const float* __restrict__ src, _Float16* __restrict__ dst, int n)
{
    int i = (blockIdx.x * 256 + threadIdx.x) * 4;
    if (i < n) {
#pragma unroll
        for (int j = 0; j < 4; ++j) dst[i + j] = (_Float16)src[i + j];
    }
}

// Gather (targets ++ neighbors) rows of x -> f16 [17408, 256]
__global__ __launch_bounds__(256) void gather_rows_f16_kernel(
    const float* __restrict__ x, const int* __restrict__ tgt,
    const int* __restrict__ nbr, _Float16* __restrict__ out)
{
    const int r = blockIdx.x;
    const int node = (r < 1024) ? tgt[r] : nbr[r - 1024];
    out[(long)r * 256 + threadIdx.x] = (_Float16)x[(long)node * 256 + threadIdx.x];
}

// ---------------------------------------------------------------------------
// Generic f16 WMMA GEMM:  C[r,n] = act( sum_k A[r,k] * W[n,k] + bias[n] )
// A: f16 [rows, lda], W: f16 [N,K] row-major, C: f16 [rows, ldc]
// Block: 256 threads = 8 waves; wave owns a 16x64 strip. The 64-col x 256-k
// B panel (shared by all 8 waves) is staged via GLOBAL_LOAD_ASYNC_TO_LDS_B128
// (ASYNCcnt), then the k-loop runs barrier-free from LDS.
// grid.x = rows/128, grid.y = cols/64, grid.z batches (heads) via strides.
// ---------------------------------------------------------------------------
__global__ __launch_bounds__(256) void gemm_f16_kernel(
    const _Float16* __restrict__ A, int lda, long strideAz,
    const _Float16* __restrict__ W, int K, long strideWz,
    const float* __restrict__ bias, long strideBz,
    _Float16* __restrict__ C, int ldc, long strideCz,
    int act)
{
    extern __shared__ _Float16 lds_b[];     // [64][PSTRIDE]

    const int z = blockIdx.z;
    A += z * strideAz;
    W += z * strideWz;
    if (bias) bias += z * strideBz;
    C += z * strideCz;

    const int lane    = threadIdx.x & 31;
    const int wave    = threadIdx.x >> 5;
    const int l16     = lane & 15;
    const int halfSel = lane >> 4;          // 0 or 1
    const int row0    = blockIdx.x * 128 + wave * 16;
    const int col0    = blockIdx.y * 64;

    // A fragment K slots per ISA layout: lanes<16 -> K 0..7,16..23 ; lanes>=16 -> K 8..15,24..31
    const int kSelA  = halfSel * 8;
    // B fragment K slots (match A): lanes<16 -> K 0..7,16..23 ; lanes>=16 -> K 8..15,24..31
    const int kBaseB = halfSel * 8;

    v8f acc[4] = {};
    const _Float16* arow = A + (long)(row0 + l16) * lda + kSelA;

    for (int kp = 0; kp < K; kp += PANEL_K) {
        const int kpLen = (K - kp) < PANEL_K ? (K - kp) : PANEL_K;
        const int segs  = kpLen >> 3;       // v8h chunks per W row in panel

        if (kp) __syncthreads();            // previous panel fully consumed
        // cooperative async global->LDS stage of the B panel
        for (int idx = threadIdx.x; idx < 64 * segs; idx += 256) {
            int n   = idx / segs;
            int sg  = (idx - n * segs) * 8;
            const _Float16* src = W + (long)(col0 + n) * K + kp + sg;
            unsigned ldsOff = (unsigned)(uintptr_t)(lds_b + n * PSTRIDE + sg);
            asm volatile("global_load_async_to_lds_b128 %0, %1, off"
                         :: "v"(ldsOff), "v"((unsigned long long)(uintptr_t)src)
                         : "memory");
        }
        asm volatile("s_wait_asynccnt 0x0" ::: "memory");
        __syncthreads();                    // panel visible to all waves

        for (int k0 = 0; k0 < kpLen; k0 += 32) {
            __builtin_prefetch(arow + kp + k0 + 64, 0, 1);  // speculative A prefetch

            v8h a0 = *(const v8h*)(arow + kp + k0);
            v8h a1 = *(const v8h*)(arow + kp + k0 + 16);
            v16h a;
#pragma unroll
            for (int j = 0; j < 8; ++j) { a[j] = a0[j]; a[8 + j] = a1[j]; }

            // load all 4 B fragments into independent regs, then 4 back-to-back WMMAs
            v16h bfr[4];
#pragma unroll
            for (int t = 0; t < 4; ++t) {
                const _Float16* bsrc = lds_b + (t * 16 + l16) * PSTRIDE + k0 + kBaseB;
                v8h b0 = *(const v8h*)(bsrc);
                v8h b1 = *(const v8h*)(bsrc + 16);
#pragma unroll
                for (int j = 0; j < 8; ++j) { bfr[t][j] = b0[j]; bfr[t][8 + j] = b1[j]; }
            }
#pragma unroll
            for (int t = 0; t < 4; ++t) {
                acc[t] = __builtin_amdgcn_wmma_f32_16x16x32_f16(
                    false, a, false, bfr[t], (short)0, acc[t], false, false);
            }
        }
    }

#pragma unroll
    for (int t = 0; t < 4; ++t) {
        int col = col0 + t * 16 + l16;
        float bs = bias ? bias[col] : 0.f;
#pragma unroll
        for (int vg = 0; vg < 8; ++vg) {
            int row = row0 + vg + halfSel * 8;
            float v = acc[t][vg] + bs;
            C[(long)row * ldc + col] = (_Float16)act_apply(v, act);
        }
    }
}

// ---------------------------------------------------------------------------
// Stage 5: att[m] = <h1[b,m,:], wh2[b,:]>, softmax over m=0..15,
//          h[o] = sum_m lrelu(vh[b,m,o]*e[m]),  out = (x_sh + h)*0.5
// grid (1024, 4), block 256. One (b, head) per block.
// ---------------------------------------------------------------------------
__global__ __launch_bounds__(256) void attn_combine_kernel(
    const _Float16* __restrict__ h1,    // [4,16384,128]
    const _Float16* __restrict__ wh2,   // [4,1024,128]
    const _Float16* __restrict__ vh,    // [4,16384,128]
    const _Float16* __restrict__ xy,    // [17408,512] rows 0..1023 = targets
    _Float16* __restrict__ outp)        // out_all + s*1024*512 -> [1024,512]
{
    const int b = blockIdx.x, h = blockIdx.y;
    const int lane = threadIdx.x & 31, wave = threadIdx.x >> 5;
    __shared__ float s_att[16];

    const _Float16* w2 = wh2 + ((long)h * 1024 + b) * 128;
#pragma unroll
    for (int mi = 0; mi < 2; ++mi) {
        int m = wave * 2 + mi;
        const _Float16* h1r = h1 + ((long)h * 16384 + b * 16 + m) * 128;
        float acc = 0.f;
#pragma unroll
        for (int j = 0; j < 4; ++j) {
            int idx = lane * 4 + j;
            acc += (float)h1r[idx] * (float)w2[idx];
        }
#pragma unroll
        for (int off = 16; off > 0; off >>= 1) acc += __shfl_xor(acc, off, 32);
        if (lane == 0) s_att[m] = acc;
    }
    __syncthreads();

    float e[16], mx = -1e30f, sum = 0.f;
#pragma unroll
    for (int m = 0; m < 16; ++m) mx = fmaxf(mx, s_att[m]);
#pragma unroll
    for (int m = 0; m < 16; ++m) { e[m] = __expf(s_att[m] - mx); sum += e[m]; }
    float inv = 1.f / sum;

    const int o = threadIdx.x;
    if (o < 128) {
        float acc = 0.f;
        const _Float16* vr = vh + ((long)h * 16384 + b * 16) * 128 + o;
#pragma unroll
        for (int m = 0; m < 16; ++m) {
            float v = (float)vr[(long)m * 128] * (e[m] * inv);
            acc += v >= 0.f ? v : LRELU_ALPHA * v;
        }
        float xs = (float)xy[(long)b * 512 + h * 128 + o];
        outp[(long)b * 512 + h * 128 + o] = (_Float16)((xs + acc) * 0.5f);
    }
}

// Pack z for metapath ki: zbuf[(b*P+p), :] = out_all[s0+p, b, :]
__global__ void pack_z_kernel(const _Float16* __restrict__ out_all,
                              _Float16* __restrict__ zbuf, int s0, int P)
{
    int r = blockIdx.x, b = r / P, p = r % P;
    const _Float16* src = out_all + ((long)(s0 + p) * 1024 + b) * 512;
    _Float16* dst = zbuf + (long)r * 512;
    for (int d = threadIdx.x; d < 512; d += 256) dst[d] = src[d];
}

// Semantic attention tail: s[p] = <w[b,p,:], p2>, softmax over p, weighted sum of z rows.
__global__ __launch_bounds__(256) void sem_score_kernel(
    const _Float16* __restrict__ w,     // [B*P, 128] (tanh already applied)
    const float* __restrict__ p2,       // [128]
    const _Float16* __restrict__ z,     // [B*P, 512]
    void* __restrict__ outp, int out_f32,
    long outRowStride, long outOffset, int P)
{
    const int b = blockIdx.x;
    const int lane = threadIdx.x & 31, wave = threadIdx.x >> 5;
    __shared__ float s_s[4];

    if (wave < P) {
        const _Float16* wr = w + ((long)b * P + wave) * 128;
        float acc = 0.f;
#pragma unroll
        for (int j = 0; j < 4; ++j) {
            int i = lane * 4 + j;
            acc += (float)wr[i] * p2[i];
        }
#pragma unroll
        for (int off = 16; off > 0; off >>= 1) acc += __shfl_xor(acc, off, 32);
        if (lane == 0) s_s[wave] = acc;
    }
    __syncthreads();

    float beta[3], mx = -1e30f, sum = 0.f;
    for (int p = 0; p < P; ++p) mx = fmaxf(mx, s_s[p]);
    for (int p = 0; p < P; ++p) { beta[p] = __expf(s_s[p] - mx); sum += beta[p]; }
    for (int p = 0; p < P; ++p) beta[p] /= sum;

    for (int d = threadIdx.x; d < 512; d += 256) {
        float acc = 0.f;
        for (int p = 0; p < P; ++p)
            acc += beta[p] * (float)z[((long)b * P + p) * 512 + d];
        if (out_f32) ((float*)outp)[(long)b * outRowStride + outOffset + d] = acc;
        else ((_Float16*)outp)[(long)b * outRowStride + outOffset + d] = (_Float16)acc;
    }
}

// ---------------------------------------------------------------------------
extern "C" void kernel_launch(void* const* d_in, const int* in_sizes, int n_in,
                              void* d_out, int out_size, void* d_ws, size_t ws_size,
                              hipStream_t stream) {
    const float* x        = (const float*)d_in[0];
    const int*   tgt      = (const int*)d_in[1];
    const int*   nbr      = (const int*)d_in[2];
    const float* fc_w     = (const float*)d_in[3];
    const float* fc_b     = (const float*)d_in[4];
    const float* q_w      = (const float*)d_in[5];
    const float* q_b      = (const float*)d_in[6];
    const float* k_w      = (const float*)d_in[7];
    const float* k_b      = (const float*)d_in[8];
    const float* v_w      = (const float*)d_in[9];
    const float* v_b      = (const float*)d_in[10];
    const float* att_W    = (const float*)d_in[11];
    const float* a1_w     = (const float*)d_in[12];
    const float* a1_b     = (const float*)d_in[13];
    const float* a2_w     = (const float*)d_in[14];
    const float* a2_b     = (const float*)d_in[15];
    const float* ip1_w    = (const float*)d_in[16];
    const float* ip1_b    = (const float*)d_in[17];
    const float* ip2_w    = (const float*)d_in[18];
    const float* bp1_w    = (const float*)d_in[19];
    const float* bp1_b    = (const float*)d_in[20];
    const float* bp2_w    = (const float*)d_in[21];

    char* ws = (char*)d_ws;
    size_t off = 0;
    auto alloc = [&](size_t bytes) -> void* {
        void* p = ws + off;
        off += (bytes + 255) & ~(size_t)255;
        return p;
    };
    _Float16* xy      = (_Float16*)alloc(17408UL * 512 * 2);   // fc output
    _Float16* xr      = (_Float16*)alloc(17408UL * 256 * 2);   // gathered f16 rows
    _Float16* qh      = (_Float16*)alloc(4UL * 1024 * 128 * 2);
    _Float16* kh      = (_Float16*)alloc(4UL * 16384 * 128 * 2);
    _Float16* vh      = (_Float16*)alloc(4UL * 16384 * 128 * 2);
    _Float16* h1      = (_Float16*)alloc(4UL * 16384 * 128 * 2);
    _Float16* h2      = (_Float16*)alloc(4UL * 1024 * 128 * 2);
    _Float16* wh2     = (_Float16*)alloc(4UL * 1024 * 128 * 2);
    _Float16* out_all = (_Float16*)alloc(5UL * 1024 * 512 * 2);
    _Float16* zbuf    = (_Float16*)alloc(3UL * 1024 * 512 * 2);
    _Float16* w_se    = (_Float16*)alloc(3UL * 1024 * 128 * 2);
    _Float16* zstk    = (_Float16*)alloc(2UL * 1024 * 512 * 2);
    // f16 weight copies
    _Float16* fcw16  = (_Float16*)alloc(655360UL * 2);
    _Float16* qw16   = (_Float16*)alloc(327680UL * 2);
    _Float16* kw16   = (_Float16*)alloc(327680UL * 2);
    _Float16* vw16   = (_Float16*)alloc(327680UL * 2);
    _Float16* aW16   = (_Float16*)alloc(131072UL * 2);
    _Float16* a1w16  = (_Float16*)alloc(131072UL * 2);
    _Float16* a2w16  = (_Float16*)alloc(131072UL * 2);
    _Float16* ip1w16 = (_Float16*)alloc(131072UL * 2);
    _Float16* bp1w16 = (_Float16*)alloc(65536UL * 2);
    (void)ws_size; (void)in_sizes; (void)n_in; (void)out_size;

    auto conv = [&](const float* s, _Float16* d, int n) {
        f32_to_f16_kernel<<<dim3((n / 4 + 255) / 256, 1, 1), 256, 0, stream>>>(s, d, n);
    };
    conv(fc_w, fcw16, 655360);
    conv(q_w, qw16, 327680);
    conv(k_w, kw16, 327680);
    conv(v_w, vw16, 327680);
    conv(att_W, aW16, 131072);
    conv(a1_w, a1w16, 131072);
    conv(a2_w, a2w16, 131072);
    conv(ip1_w, ip1w16, 131072);
    conv(bp1_w, bp1w16, 65536);

    static const int K_OF_S[5] = {0, 0, 1, 1, 1};
    const size_t SH = GEMM_LDS_BYTES;

    for (int s = 0; s < 5; ++s) {
        const int kk = K_OF_S[s];
        // Stage 1a: gather + convert activation rows -> xr [17408, 256] f16
        gather_rows_f16_kernel<<<dim3(17408, 1, 1), 256, 0, stream>>>(
            x, tgt, nbr + (long)s * 1024 * 16, xr);
        // Stage 1b: fc GEMM (all heads as one 512-wide N) + lrelu -> xy
        gemm_f16_kernel<<<dim3(136, 8, 1), 256, SH, stream>>>(
            xr, 256, 0L,
            fcw16 + (long)s * 512 * 256, 256, 0L,
            fc_b + (long)s * 512, 0L,
            xy, 512, 0L, 1);
        // q: rows 0..1023 of xy, per head
        gemm_f16_kernel<<<dim3(8, 2, 4), 256, SH, stream>>>(
            xy, 512, 128L,
            qw16 + (long)s * 4 * 16384, 128, 16384L,
            q_b + (long)s * 4 * 128, 128L,
            qh, 128, 1024L * 128, 0);
        // k, v: rows 1024..17407 of xy
        gemm_f16_kernel<<<dim3(128, 2, 4), 256, SH, stream>>>(
            xy + 1024L * 512, 512, 128L,
            kw16 + (long)s * 4 * 16384, 128, 16384L,
            k_b + (long)s * 4 * 128, 128L,
            kh, 128, 16384L * 128, 0);
        gemm_f16_kernel<<<dim3(128, 2, 4), 256, SH, stream>>>(
            xy + 1024L * 512, 512, 128L,
            vw16 + (long)s * 4 * 16384, 128, 16384L,
            v_b + (long)s * 4 * 128, 128L,
            vh, 128, 16384L * 128, 0);
        // h1 = lrelu(kh @ a1w^T + a1b)
        gemm_f16_kernel<<<dim3(128, 2, 4), 256, SH, stream>>>(
            kh, 128, 16384L * 128,
            a1w16 + (long)kk * 4 * 16384, 128, 16384L,
            a1_b + (long)kk * 4 * 128, 128L,
            h1, 128, 16384L * 128, 1);
        // h2 = lrelu(qh @ a2w^T + a2b)
        gemm_f16_kernel<<<dim3(8, 2, 4), 256, SH, stream>>>(
            qh, 128, 1024L * 128,
            a2w16 + (long)kk * 4 * 16384, 128, 16384L,
            a2_b + (long)kk * 4 * 128, 128L,
            h2, 128, 1024L * 128, 1);
        // wh2 = h2 @ aW^T  (wh2[b,i] = sum_j aW[i,j] h2[b,j])
        gemm_f16_kernel<<<dim3(8, 2, 4), 256, SH, stream>>>(
            h2, 128, 1024L * 128,
            aW16 + (long)kk * 4 * 16384, 128, 16384L,
            (const float*)nullptr, 0L,
            wh2, 128, 1024L * 128, 0);
        // softmax attention + combine -> out_all[s]
        attn_combine_kernel<<<dim3(1024, 4, 1), 256, 0, stream>>>(
            h1, wh2, vh, xy, out_all + (long)s * 1024 * 512);
    }

    // Inner semantic attention per metapath
    static const int S0[2] = {0, 2};
    static const int PP[2] = {2, 3};
    for (int ki = 0; ki < 2; ++ki) {
        const int s0 = S0[ki], P = PP[ki];
        pack_z_kernel<<<dim3(1024 * P, 1, 1), 256, 0, stream>>>(out_all, zbuf, s0, P);
        gemm_f16_kernel<<<dim3(1024 * P / 128, 2, 1), 256, SH, stream>>>(
            zbuf, 512, 0L,
            ip1w16 + (long)ki * 128 * 512, 512, 0L,
            ip1_b + (long)ki * 128, 0L,
            w_se, 128, 0L, 2);
        sem_score_kernel<<<dim3(1024, 1, 1), 256, 0, stream>>>(
            w_se, ip2_w + (long)ki * 128, zbuf,
            (void*)zstk, 0, 1024L, (long)ki * 512, P);
    }

    // Between-metapath semantic attention -> final f32 output [1024, 512]
    gemm_f16_kernel<<<dim3(16, 2, 1), 256, SH, stream>>>(
        zstk, 512, 0L,
        bp1w16, 512, 0L,
        bp1_b, 0L,
        w_se, 128, 0L, 2);
    sem_score_kernel<<<dim3(1024, 1, 1), 256, 0, stream>>>(
        w_se, bp2_w, zstk, d_out, 1, 512L, 0L, 2);
}
